// MotifConnectionsLayer_11982958755988
// MI455X (gfx1250) — matrix operand order repaired and checked by reading, not compiled, not run on hardware
//
#include <hip/hip_runtime.h>

#define B_ 32
#define F_ 1024
#define L_ 1000
#define N_ 2048

#define LB 64    // l-tile per workgroup (4 x 16 WMMA subtiles)
#define NB 256   // n-tile per workgroup (8 waves x 2 strips x 16 rows)
#define KB 32    // K (f) step = bf16 WMMA depth

#define BSTRIDE 40  // bf16 per l-row in LDS (80 B: 16B-aligned chunks, bank-spread)
#define ASTRIDE 40  // bf16 per n-row in LDS

typedef __attribute__((ext_vector_type(16))) __bf16 v16bf;
typedef __attribute__((ext_vector_type(8)))  __bf16 v8bf;
typedef __attribute__((ext_vector_type(8)))  float  v8f;

__launch_bounds__(256)
__global__ void motif_gemm_bf16x2(const float* __restrict__ X,
                                  const float* __restrict__ C,
                                  const float* __restrict__ W,
                                  float* __restrict__ out) {
    __shared__ __bf16 lds_a  [NB * ASTRIDE];  // bf16 C^T tile, [n][k]   (20.0 KB)
    __shared__ __bf16 lds_bhi[LB * BSTRIDE];  // bf16 hi(X) tile, [l][k] ( 5.0 KB)
    __shared__ __bf16 lds_blo[LB * BSTRIDE];  // bf16 lo(X) tile, [l][k] ( 5.0 KB)

    const int l0   = blockIdx.x * LB;
    const int n0   = blockIdx.y * NB;
    const int b    = blockIdx.z;
    const int tid  = threadIdx.x;
    const int wave = tid >> 5;            // 0..7
    const int lane = tid & 31;

    const float* Xb = X + (size_t)b * F_ * L_;

    v8f acc0[4] = {{}, {}, {}, {}};       // strip 0 (n rows wave*16 + [0,16))
    v8f acc1[4] = {{}, {}, {}, {}};       // strip 1 (n rows 128 + wave*16 + [0,16))

    // Operand-gather lane roles (wave32); M/N per CDNA5 ISA, K order contiguous (this code's choice):
    const int mrow    = lane & 15;   // A row (M) / B column within subtile
    const int hi_half = lane >> 4;   // 0: lanes 0-15, 1: lanes 16-31
    const int a_k0    = hi_half * 16; // A: 16 consecutive K (0..15 vs 16..31)
    const int b_k0    = hi_half * 16; // B: 16 consecutive K (0..15 vs 16..31)

    // Staging roles (constant across K-steps):
    const int xl    = tid & 63;       // X: l within tile (lanes sweep l -> coalesced)
    const int xkgrp = tid >> 6;       // X: which group of 8 consecutive k
    const int an    = tid & 127;      // C: n within tile (lanes sweep n -> coalesced)
    const int akset = (tid >> 7) * 16;// C: which 16 consecutive k

    // Branch-free edge handling: clamp the address. Column l of B only feeds
    // column l of D, and stores mask l >= L, so duplicated edge data is inert.
    const int glx = min(l0 + xl, L_ - 1);

    // ---- register double-buffer: raw f32 prefetch of one K-step ----
    float xr[8];
    float cr0[16], cr1[16];   // C rows an and an+128
    {   // preload K-step 0
        const float* xsrc = Xb + (size_t)(xkgrp * 8) * L_ + glx;
        #pragma unroll
        for (int kk = 0; kk < 8; ++kk) xr[kk] = xsrc[(size_t)kk * L_];
        const float* csrc = C + (size_t)akset * N_ + (n0 + an);
        #pragma unroll
        for (int kk = 0; kk < 16; ++kk) cr0[kk] = csrc[(size_t)kk * N_];
        #pragma unroll
        for (int kk = 0; kk < 16; ++kk) cr1[kk] = csrc[(size_t)kk * N_ + 128];
    }

    for (int f0 = 0; f0 < F_; f0 += KB) {
        // ---- convert prefetched regs, stage to LDS (one-time hi/lo split) ----
        {
            v8bf hv, lv;
            #pragma unroll
            for (int kk = 0; kk < 8; ++kk) {
                const float  x = xr[kk];
                const __bf16 h = (__bf16)x;
                hv[kk] = h;
                lv[kk] = (__bf16)(x - (float)h);
            }
            *reinterpret_cast<v8bf*>(&lds_bhi[xl * BSTRIDE + xkgrp * 8]) = hv;
            *reinterpret_cast<v8bf*>(&lds_blo[xl * BSTRIDE + xkgrp * 8]) = lv;

            v8bf av0, av1, av2, av3;
            #pragma unroll
            for (int kk = 0; kk < 8; ++kk) av0[kk] = (__bf16)cr0[kk];
            #pragma unroll
            for (int kk = 0; kk < 8; ++kk) av1[kk] = (__bf16)cr0[kk + 8];
            #pragma unroll
            for (int kk = 0; kk < 8; ++kk) av2[kk] = (__bf16)cr1[kk];
            #pragma unroll
            for (int kk = 0; kk < 8; ++kk) av3[kk] = (__bf16)cr1[kk + 8];
            __bf16* arow0 = &lds_a[an * ASTRIDE + akset];
            __bf16* arow1 = &lds_a[(an + 128) * ASTRIDE + akset];
            *reinterpret_cast<v8bf*>(arow0)     = av0;
            *reinterpret_cast<v8bf*>(arow0 + 8) = av1;
            *reinterpret_cast<v8bf*>(arow1)     = av2;
            *reinterpret_cast<v8bf*>(arow1 + 8) = av3;
        }
        __syncthreads();   // tile visible to all waves

        // ---- issue next K-step's global loads; they fly under the WMMAs ----
        if (f0 + KB < F_) {
            const float* xsrc = Xb + (size_t)(f0 + KB + xkgrp * 8) * L_ + glx;
            #pragma unroll
            for (int kk = 0; kk < 8; ++kk) xr[kk] = xsrc[(size_t)kk * L_];
            const float* csrc = C + (size_t)(f0 + KB + akset) * N_ + (n0 + an);
            #pragma unroll
            for (int kk = 0; kk < 16; ++kk) cr0[kk] = csrc[(size_t)kk * N_];
            #pragma unroll
            for (int kk = 0; kk < 16; ++kk) cr1[kk] = csrc[(size_t)kk * N_ + 128];
        }

        // ---- gather BOTH A operands (16x32 bf16) for this wave ----
        v16bf a0, a1;
        {
            const __bf16* ar0 = &lds_a[(wave * 16 + mrow) * ASTRIDE];
            const __bf16* ar1 = ar0 + 128 * ASTRIDE;
            const v8bf c0 = *reinterpret_cast<const v8bf*>(&ar0[a_k0]);
            const v8bf c1 = *reinterpret_cast<const v8bf*>(&ar0[a_k0 + 8]);
            const v8bf c2 = *reinterpret_cast<const v8bf*>(&ar1[a_k0]);
            const v8bf c3 = *reinterpret_cast<const v8bf*>(&ar1[a_k0 + 8]);
            #pragma unroll
            for (int i = 0; i < 8; ++i) {
                a0[i] = c0[i]; a0[8 + i] = c1[i];
                a1[i] = c2[i]; a1[8 + i] = c3[i];
            }
        }

        // ---- per l-subtile: one B pair feeds BOTH A strips (4 WMMAs) ----
        #pragma unroll
        for (int sub = 0; sub < 4; ++sub) {
            const int brow = (sub * 16 + mrow) * BSTRIDE + b_k0;
            const v16bf bh = *reinterpret_cast<const v16bf*>(&lds_bhi[brow]);
            const v16bf bl = *reinterpret_cast<const v16bf*>(&lds_blo[brow]);
            acc0[sub] = __builtin_amdgcn_wmma_f32_16x16x32_bf16(
                            false, a0, false, bh, (short)0, acc0[sub], false, false);
            acc0[sub] = __builtin_amdgcn_wmma_f32_16x16x32_bf16(
                            false, a0, false, bl, (short)0, acc0[sub], false, false);
            acc1[sub] = __builtin_amdgcn_wmma_f32_16x16x32_bf16(
                            false, a1, false, bh, (short)0, acc1[sub], false, false);
            acc1[sub] = __builtin_amdgcn_wmma_f32_16x16x32_bf16(
                            false, a1, false, bl, (short)0, acc1[sub], false, false);
        }
        __syncthreads();   // all reads done; next iteration may overwrite LDS
    }

    // ---- epilogue: fold W[n] (fp32) and store both strips ----
    // D layout: VGPR r -> M = r + 8*hi_half ; N (=l col) = lane & 15
    #pragma unroll
    for (int strip = 0; strip < 2; ++strip) {
        const int nrow_base = n0 + strip * 128 + wave * 16 + hi_half * 8;
        const int lcol      = mrow;
        float wv[8];
        #pragma unroll
        for (int r = 0; r < 8; ++r) wv[r] = W[nrow_base + r];

        const v8f* accs = strip ? acc1 : acc0;
        #pragma unroll
        for (int sub = 0; sub < 4; ++sub) {
            const int gl = l0 + sub * 16 + lcol;
            if (gl < L_) {
                float* o = out + ((size_t)b * N_ + nrow_base) * L_ + gl;
                #pragma unroll
                for (int r = 0; r < 8; ++r) o[(size_t)r * L_] = accs[sub][r] * wv[r];
            }
        }
    }
}

extern "C" void kernel_launch(void* const* d_in, const int* in_sizes, int n_in,
                              void* d_out, int out_size, void* d_ws, size_t ws_size,
                              hipStream_t stream) {
    const float* X = (const float*)d_in[0];
    const float* C = (const float*)d_in[1];
    const float* W = (const float*)d_in[2];
    float* out = (float*)d_out;

    dim3 grid((L_ + LB - 1) / LB,  // 16 l-tiles (last partial: 960..999 valid)
              N_ / NB,             // 8 n-tiles
              B_);                 // 32 batches
    motif_gemm_bf16x2<<<grid, 256, 0, stream>>>(X, C, W, out);
}
